// DrugRepresentationModule_53326313947260
// MI455X (gfx1250) — matrix operand order repaired and checked by reading, not compiled, not run on hardware
//
#include <hip/hip_runtime.h>
#include <hip/hip_bf16.h>

typedef __attribute__((ext_vector_type(16))) __bf16 v16bf;
typedef __attribute__((ext_vector_type(8)))  float  v8f;

#define N_NODES   50000
#define N_EDGES   800000
#define ATOM_DIM  78
#define HIDDEN    128
#define OUTDIM    100
#define N_GRAPHS  500
#define N_LAYERS  3
// 1/sqrt(1 + 1e-5) computed on device via rsqrtf(1.00001f)

// ---------------------------------------------------------------------------
// Weight prep: Wt[m][n][k] = bf16( W[m's layer][k][n] * g[n]*rsqrt(1+eps) )
// m = layer*2 + (0 for W1, 1 for W2).  Transposed + BN-folded, bf16.
// ---------------------------------------------------------------------------
__global__ void prep_w_kernel(const float* __restrict__ W1, const float* __restrict__ g1,
                              const float* __restrict__ W2, const float* __restrict__ g2,
                              __bf16* __restrict__ Wt) {
    int idx = blockIdx.x * blockDim.x + threadIdx.x;       // 6*128*128
    if (idx >= 6 * HIDDEN * HIDDEN) return;
    int m = idx >> 14;                 // which matrix (0..5)
    int r = idx & 16383;
    int n = r >> 7;                    // output column
    int k = r & 127;                   // k index
    int layer = m >> 1;
    const float* W = (m & 1) ? W2 : W1;
    const float* g = (m & 1) ? g2 : g1;
    float s = g[layer * HIDDEN + n] * rsqrtf(1.00001f);
    Wt[idx] = (__bf16)(W[layer * HIDDEN * HIDDEN + k * HIDDEN + n] * s);
}

__global__ void prep_b_kernel(const float* __restrict__ b1, const float* __restrict__ g1,
                              const float* __restrict__ be1,
                              const float* __restrict__ b2, const float* __restrict__ g2,
                              const float* __restrict__ be2,
                              float* __restrict__ biasf) {
    int idx = blockIdx.x * blockDim.x + threadIdx.x;       // 6*128
    if (idx >= 6 * HIDDEN) return;
    int m = idx >> 7;
    int n = idx & 127;
    int layer = m >> 1;
    const float* b  = (m & 1) ? b2  : b1;
    const float* g  = (m & 1) ? g2  : g1;
    const float* be = (m & 1) ? be2 : be1;
    float s = g[layer * HIDDEN + n] * rsqrtf(1.00001f);
    biasf[idx] = b[layer * HIDDEN + n] * s + be[layer * HIDDEN + n];
}

// ---------------------------------------------------------------------------
// Embedding: h[node][c] = sum_k x[node][k] * W_emb[k][c] + b_emb[c]
// one block per node, 128 threads, x row staged in LDS.
// ---------------------------------------------------------------------------
__global__ __launch_bounds__(HIDDEN)
void embed_kernel(const float* __restrict__ x, const float* __restrict__ W,
                  const float* __restrict__ b, float* __restrict__ h) {
    __shared__ float xs[ATOM_DIM];
    int node = blockIdx.x;
    int t = threadIdx.x;
    if (t < ATOM_DIM) xs[t] = x[(size_t)node * ATOM_DIM + t];
    __syncthreads();
    float acc = b[t];
    #pragma unroll 6
    for (int k = 0; k < ATOM_DIM; ++k) acc += xs[k] * W[k * HIDDEN + t];
    h[(size_t)node * HIDDEN + t] = acc;
}

// ---------------------------------------------------------------------------
// Edge aggregation: agg[dst] += h[src]  (one thread per edge x float4)
// ---------------------------------------------------------------------------
__global__ void edge_agg_kernel(const int* __restrict__ src, const int* __restrict__ dst,
                                const float* __restrict__ h, float* __restrict__ agg) {
    long gid = (long)blockIdx.x * blockDim.x + threadIdx.x;
    int e = (int)(gid >> 5);
    int f = (int)(gid & 31);
    if (e >= N_EDGES) return;
    int s = src[e];
    int d = dst[e];
    float4 v = ((const float4*)h)[(size_t)s * 32 + f];
    float* o = agg + (size_t)d * HIDDEN + f * 4;
    atomicAdd(o + 0, v.x);
    atomicAdd(o + 1, v.y);
    atomicAdd(o + 2, v.z);
    atomicAdd(o + 3, v.w);
}

// ---------------------------------------------------------------------------
// Combine: z_bf16 = bf16( (1+eps[layer])*h + agg )
// ---------------------------------------------------------------------------
__global__ void combine_kernel(const float* __restrict__ h, const float* __restrict__ agg,
                               const float* __restrict__ eps, int layer,
                               __bf16* __restrict__ z) {
    size_t i = (size_t)blockIdx.x * blockDim.x + threadIdx.x;
    if (i >= (size_t)N_NODES * HIDDEN) return;
    float e = 1.0f + eps[layer];
    z[i] = (__bf16)(e * h[i] + agg[i]);
}

// ---------------------------------------------------------------------------
// Node GEMM via v_wmma_f32_16x16x32_bf16:
//   Out[N,128] = relu( A[N,128](bf16) @ W'(bf16, BN-folded) + bias' )
// Weights pre-transposed to [n][k] so the LDS fill is a straight copy.
// Block = 4 waves; each wave owns one 16-row strip, 8 N-tiles, 4 K-chunks.
// ---------------------------------------------------------------------------
#define WPITCH 136   // halves per LDS row: 272B = 17*16B -> b128 aligned, no bank conflicts

template <bool OUT_BF16>
__global__ __launch_bounds__(128)
void gemm_node_kernel(const __bf16* __restrict__ A, const __bf16* __restrict__ Wt,
                      const float* __restrict__ bias, void* __restrict__ Out, int nrows) {
    __shared__ __bf16 wlds[HIDDEN * WPITCH];
    const int t = threadIdx.x;

    // cooperative LDS fill: 128x128 bf16, 8 halves (16B) per iteration
    for (int i = t; i < HIDDEN * 16; i += 128) {
        int n = i >> 4, kg = i & 15;
        *(uint4*)&wlds[n * WPITCH + kg * 8] = *(const uint4*)&Wt[n * HIDDEN + kg * 8];
    }
    __syncthreads();

    const int wave = t >> 5;
    const int lane = t & 31;
    const int tile = blockIdx.x * 4 + wave;          // 16-row strip index
    if (tile * 16 >= nrows) return;                  // wave-uniform: EXEC stays all-ones
    const int lrow = lane & 15;
    const int hi   = lane >> 4;                      // half-wave selector
    const int row  = tile * 16 + lrow;               // A row for this lane (both halves)

    v8f acc[8] = {};

    #pragma unroll
    for (int kc = 0; kc < 4; ++kc) {
        // A fragment (ISA 7.12.2, 16-bit A 16x32):
        //  lane<16: halves0-7 => K=kc*32+0..7,  halves8-15 => K=kc*32+16..23
        //  lane>=16: halves0-7 => K=kc*32+8..15, halves8-15 => K=kc*32+24..31
        union { uint4 u[2]; v16bf v; } af;
        const __bf16* arow = A + (size_t)row * HIDDEN + kc * 32 + hi * 8;
        af.u[0] = *(const uint4*)(arow);
        af.u[1] = *(const uint4*)(arow + 16);

        #pragma unroll
        for (int nt = 0; nt < 8; ++nt) {
            // B fragment (32x16): lane N = lane%16; K = kc*32 + hi*8 + 0..7, kc*32 + 16 + hi*8 + 0..7
            union { uint4 u[2]; v16bf v; } bf_;
            const __bf16* bp = &wlds[(nt * 16 + lrow) * WPITCH + kc * 32 + hi * 8];
            bf_.u[0] = *(const uint4*)(bp);
            bf_.u[1] = *(const uint4*)(bp + 16);
            acc[nt] = __builtin_amdgcn_wmma_f32_16x16x32_bf16(
                false, af.v, false, bf_.v, (short)0, acc[nt], false, false);
        }
    }

    // Epilogue: C/D layout -> VGPR r: M = r + hi*8, N = nt*16 + lane%16
    #pragma unroll
    for (int nt = 0; nt < 8; ++nt) {
        int col = nt * 16 + lrow;
        float b = bias[col];
        #pragma unroll
        for (int r = 0; r < 8; ++r) {
            float v = acc[nt][r] + b;
            v = v > 0.0f ? v : 0.0f;
            size_t o = (size_t)(tile * 16 + r + hi * 8) * HIDDEN + col;
            if (OUT_BF16) ((__bf16*)Out)[o] = (__bf16)v;
            else          ((float*)Out)[o]  = v;
        }
    }
}

// ---------------------------------------------------------------------------
// Pooling
// ---------------------------------------------------------------------------
__global__ void pool_acc_kernel(const float* __restrict__ h, const int* __restrict__ batch,
                                float* __restrict__ sums, float* __restrict__ counts) {
    long gid = (long)blockIdx.x * blockDim.x + threadIdx.x;
    int node = (int)(gid >> 5);
    int f = (int)(gid & 31);
    if (node >= N_NODES) return;
    int g = batch[node];
    float4 v = ((const float4*)h)[(size_t)node * 32 + f];
    float* o = sums + (size_t)g * HIDDEN + f * 4;
    atomicAdd(o + 0, v.x);
    atomicAdd(o + 1, v.y);
    atomicAdd(o + 2, v.z);
    atomicAdd(o + 3, v.w);
    if (f == 0) atomicAdd(&counts[g], 1.0f);
}

__global__ void pool_fin_kernel(const float* __restrict__ sums, const float* __restrict__ counts,
                                float* __restrict__ gmat) {
    int idx = blockIdx.x * blockDim.x + threadIdx.x;   // 500*128
    if (idx >= N_GRAPHS * HIDDEN) return;
    int gi = idx >> 7;
    float c = counts[gi];
    c = c < 1.0f ? 1.0f : c;
    gmat[idx] = sums[idx] / c;
}

// ---------------------------------------------------------------------------
// Head: tiny GEMMs (500 rows) with fused BN + ReLU
// ---------------------------------------------------------------------------
__global__ void fc1_kernel(const float* __restrict__ gmat, const float* __restrict__ Wf1,
                           const float* __restrict__ bf1, const float* __restrict__ gf1,
                           const float* __restrict__ bef1, float* __restrict__ gh) {
    int idx = blockIdx.x * blockDim.x + threadIdx.x;   // 500*128
    if (idx >= N_GRAPHS * HIDDEN) return;
    int gi = idx >> 7, n = idx & 127;
    const float* row = gmat + (size_t)gi * HIDDEN;
    float acc = 0.0f;
    #pragma unroll 8
    for (int k = 0; k < HIDDEN; ++k) acc += row[k] * Wf1[k * HIDDEN + n];
    float s = gf1[n] * rsqrtf(1.00001f);
    float v = (acc + bf1[n]) * s + bef1[n];
    gh[idx] = v > 0.0f ? v : 0.0f;
}

__global__ void fc2_kernel(const float* __restrict__ gh, const float* __restrict__ Wf2,
                           const float* __restrict__ bf2, const float* __restrict__ gf2,
                           const float* __restrict__ bef2, float* __restrict__ out) {
    int idx = blockIdx.x * blockDim.x + threadIdx.x;   // 500*100
    if (idx >= N_GRAPHS * OUTDIM) return;
    int gi = idx / OUTDIM, n = idx % OUTDIM;
    const float* row = gh + (size_t)gi * HIDDEN;
    float acc = 0.0f;
    #pragma unroll 8
    for (int k = 0; k < HIDDEN; ++k) acc += row[k] * Wf2[k * OUTDIM + n];
    float s = gf2[n] * rsqrtf(1.00001f);
    float v = (acc + bf2[n]) * s + bef2[n];
    out[idx] = v > 0.0f ? v : 0.0f;
}

// ---------------------------------------------------------------------------
// Host orchestration
// ---------------------------------------------------------------------------
static inline size_t align256(size_t x) { return (x + 255) & ~(size_t)255; }

extern "C" void kernel_launch(void* const* d_in, const int* in_sizes, int n_in,
                              void* d_out, int out_size, void* d_ws, size_t ws_size,
                              hipStream_t stream) {
    const float* x     = (const float*)d_in[0];
    const int*   edge  = (const int*)d_in[1];     // [2, N_EDGES]
    const int*   batch = (const int*)d_in[2];
    const float* W_emb = (const float*)d_in[3];
    const float* b_emb = (const float*)d_in[4];
    const float* eps   = (const float*)d_in[5];
    const float* W1    = (const float*)d_in[6];
    const float* b1    = (const float*)d_in[7];
    const float* g1    = (const float*)d_in[8];
    const float* be1   = (const float*)d_in[9];
    const float* W2    = (const float*)d_in[10];
    const float* b2    = (const float*)d_in[11];
    const float* g2    = (const float*)d_in[12];
    const float* be2   = (const float*)d_in[13];
    const float* Wf1   = (const float*)d_in[14];
    const float* bf1   = (const float*)d_in[15];
    const float* gf1   = (const float*)d_in[16];
    const float* bef1  = (const float*)d_in[17];
    const float* Wf2   = (const float*)d_in[18];
    const float* bf2   = (const float*)d_in[19];
    const float* gf2   = (const float*)d_in[20];
    const float* bef2  = (const float*)d_in[21];

    const int* src = edge;
    const int* dst = edge + N_EDGES;

    // workspace carve-up
    char* ws = (char*)d_ws;
    size_t off = 0;
    const size_t hfN = (size_t)N_NODES * HIDDEN;
    float*  h    = (float*)(ws + off);  off += align256(hfN * sizeof(float));
    float*  agg  = (float*)(ws + off);  off += align256(hfN * sizeof(float));
    __bf16* zb   = (__bf16*)(ws + off); off += align256(hfN * sizeof(__bf16));
    __bf16* tb   = (__bf16*)(ws + off); off += align256(hfN * sizeof(__bf16));
    __bf16* Wt   = (__bf16*)(ws + off); off += align256((size_t)6 * HIDDEN * HIDDEN * sizeof(__bf16));
    float*  biasf= (float*)(ws + off);  off += align256((size_t)6 * HIDDEN * sizeof(float));
    float*  sums = (float*)(ws + off);  off += align256((size_t)N_GRAPHS * HIDDEN * sizeof(float));
    float*  cnts = (float*)(ws + off);  off += align256((size_t)N_GRAPHS * sizeof(float));
    float*  gmat = (float*)(ws + off);  off += align256((size_t)N_GRAPHS * HIDDEN * sizeof(float));
    float*  gh   = (float*)(ws + off);  off += align256((size_t)N_GRAPHS * HIDDEN * sizeof(float));

    // 1) fold BN into weights/biases, transpose, convert to bf16
    prep_w_kernel<<<(6 * HIDDEN * HIDDEN + 255) / 256, 256, 0, stream>>>(W1, g1, W2, g2, Wt);
    prep_b_kernel<<<3, 256, 0, stream>>>(b1, g1, be1, b2, g2, be2, biasf);

    // 2) atom embedding
    embed_kernel<<<N_NODES, HIDDEN, 0, stream>>>(x, W_emb, b_emb, h);

    // 3) GIN layers
    const int gemm_grid = (N_NODES / 16 + 3) / 4;   // 3125 strips / 4 waves per block
    for (int layer = 0; layer < N_LAYERS; ++layer) {
        hipMemsetAsync(agg, 0, hfN * sizeof(float), stream);
        edge_agg_kernel<<<(int)(((long)N_EDGES * 32 + 255) / 256), 256, 0, stream>>>(src, dst, h, agg);
        combine_kernel<<<(int)((hfN + 255) / 256), 256, 0, stream>>>(h, agg, eps, layer, zb);
        gemm_node_kernel<true><<<gemm_grid, 128, 0, stream>>>(
            zb, Wt + (size_t)(layer * 2 + 0) * HIDDEN * HIDDEN, biasf + (layer * 2 + 0) * HIDDEN,
            (void*)tb, N_NODES);
        gemm_node_kernel<false><<<gemm_grid, 128, 0, stream>>>(
            tb, Wt + (size_t)(layer * 2 + 1) * HIDDEN * HIDDEN, biasf + (layer * 2 + 1) * HIDDEN,
            (void*)h, N_NODES);
    }

    // 4) global mean pool
    hipMemsetAsync(sums, 0, (size_t)N_GRAPHS * HIDDEN * sizeof(float), stream);
    hipMemsetAsync(cnts, 0, (size_t)N_GRAPHS * sizeof(float), stream);
    pool_acc_kernel<<<(int)(((long)N_NODES * 32 + 255) / 256), 256, 0, stream>>>(h, batch, sums, cnts);
    pool_fin_kernel<<<(N_GRAPHS * HIDDEN + 255) / 256, 256, 0, stream>>>(sums, cnts, gmat);

    // 5) head
    fc1_kernel<<<(N_GRAPHS * HIDDEN + 255) / 256, 256, 0, stream>>>(gmat, Wf1, bf1, gf1, bef1, gh);
    fc2_kernel<<<(N_GRAPHS * OUTDIM + 255) / 256, 256, 0, stream>>>(gh, Wf2, bf2, gf2, bef2, (float*)d_out);
}
